// RNNAttentionDecoder_88261577933056
// MI455X (gfx1250) — matrix operand order repaired and checked by reading, not compiled, not run on hardware
//
#include <hip/hip_runtime.h>
#include <cstddef>
#include <cstdint>

#define DM   512
#define BSZ  128
#define TSZ  128
#define CLS  8000
#define MAXLEN 80

// LDS B-tile: 64 rows x 256 K-elems (bf16), row padded by 8 u16 to dodge bank conflicts
#define KHALF 256
#define LDB   264   // 256 + 8 pad -> row stride 528B = 132 dwords, 4-bank skew per row

typedef __attribute__((ext_vector_type(16))) __bf16 v16bf;
typedef __attribute__((ext_vector_type(8)))  float  v8f;

union Frag16 {
  uint4 u[2];
  unsigned short s[16];
  v16bf v;
};

__device__ __forceinline__ unsigned short f2bf(float f) {
  union { __bf16 b; unsigned short s; } u;
  u.b = (__bf16)f;   // native hardware conversion on gfx1250
  return u.s;
}

// ---- WMMA fragment loaders (CDNA5 16x16x32 bf16 layouts, ISA 7.12.2) ----
// A 16x32 (MxK): lane L -> M=L%16; elem e<8: K=(L/16)*8+e ; e>=8: K=16+(L/16)*8+(e-8)
__device__ __forceinline__ v16bf load_a_frag(const unsigned short* A, int lda,
                                             int row, int kb, int lane) {
  const unsigned short* p = A + (size_t)row * lda + kb + ((lane >> 4) << 3);
  Frag16 f;
  f.u[0] = *(const uint4*)(p);
  f.u[1] = *(const uint4*)(p + 16);
  return f.v;
}
// Same A fragment, sourced from fp32 with native bf16 convert
__device__ __forceinline__ v16bf load_a_frag_f32(const float* A, int lda,
                                                 int row, int kb, int lane) {
  const float* p = A + (size_t)row * lda + kb + ((lane >> 4) << 3);
  v16bf v;
#pragma unroll
  for (int i = 0; i < 8; ++i) v[i] = (__bf16)p[i];
#pragma unroll
  for (int i = 0; i < 8; ++i) v[8 + i] = (__bf16)p[16 + i];
  return v;
}
// B 32x16 (KxN) fragment from LDS tile (row-major [64][LDB], row = local N)
__device__ __forceinline__ v16bf load_b_frag_lds(const unsigned short* tile,
                                                 int nloc, int kb, int lane) {
  const unsigned short* p = tile + nloc * LDB + kb + ((lane >> 4) << 3);
  Frag16 f;
  f.u[0] = *(const uint4*)(p);
  f.u[1] = *(const uint4*)(p + 16);
  return f.v;
}

__device__ __forceinline__ void wmma_pair(const v16bf& a, const v16bf& b0,
                                          const v16bf& b1, v8f& acc0, v8f& acc1) {
  acc0 = __builtin_amdgcn_wmma_f32_16x16x32_bf16(false, a, false, b0, (short)0, acc0, false, false);
  acc1 = __builtin_amdgcn_wmma_f32_16x16x32_bf16(false, a, false, b1, (short)0, acc1, false, false);
}

// Async-DMA a 64 x KHALF bf16 weight slab (rows nbase..nbase+63, K = kh..kh+255)
// into LDS via the CDNA5 async global->LDS path (ASYNCcnt-tracked, no VGPR data).
__device__ __forceinline__ void stage_B_half(const unsigned short* W, int ldw, int nbase,
                                             int kh, unsigned short* tile) {
  __syncthreads();  // prior consumers of the tile are done
  for (int i = threadIdx.x; i < 64 * (KHALF / 8); i += 256) {  // 2048 16B chunks
    int row = i >> 5, chunk = i & 31;
    const unsigned short* g = W + (size_t)(nbase + row) * ldw + kh + chunk * 8;
    unsigned lds_off = (unsigned)(uintptr_t)(tile + row * LDB + chunk * 8);
    asm volatile("global_load_async_to_lds_b128 %0, %1, off"
                 :: "v"(lds_off), "v"((unsigned long long)(uintptr_t)g)
                 : "memory");
  }
  asm volatile("s_wait_asynccnt 0x0" ::: "memory");
  __syncthreads();
}

// ---- setup kernels ----
__global__ void k_wsum(unsigned short* __restrict__ dst,
                       const float* __restrict__ Wq, const float* __restrict__ Wk) {
  int idx = blockIdx.x * 256 + threadIdx.x;  // 512*512 = 262144
  dst[idx] = f2bf(Wq[idx] + Wk[idx]);
}

__global__ void k_convert(unsigned short* __restrict__ dst, const float* __restrict__ src,
                          int rows, int src_ld, int cols) {
  long total = (long)rows * cols;
  for (long idx = (long)blockIdx.x * 256 + threadIdx.x; idx < total;
       idx += (long)gridDim.x * 256) {
    long r = idx / cols, c = idx % cols;
    dst[idx] = f2bf(src[r * src_ld + c]);
  }
}

// e[r] += sum_n tanh( (X @ Wsum^T)[r,n] + bq[n]+bk[n] ) * v[n]   (64x64 tile/block)
__global__ void __launch_bounds__(256) k_att_gemm(
    const float* __restrict__ X, const unsigned short* __restrict__ Wsum,
    const float* __restrict__ bq, const float* __restrict__ bk,
    const float* __restrict__ vvec, float* __restrict__ e) {
  __shared__ __align__(16) unsigned short tb[64 * LDB];
  __shared__ float rowsum[64];
  int lane = threadIdx.x & 31, wave = threadIdx.x >> 5;
  if (threadIdx.x < 64) rowsum[threadIdx.x] = 0.0f;
  int mbase = blockIdx.x * 64, nbase = blockIdx.y * 64;
  int mt = wave & 3, nt = (wave >> 2) * 2;
  int arow = mbase + mt * 16 + (lane & 15);
  int nl0 = nt * 16 + (lane & 15), nl1 = nl0 + 16;
  int n0 = nbase + nl0, n1 = nbase + nl1;
  v8f acc0 = {}, acc1 = {};
  for (int kh = 0; kh < DM; kh += KHALF) {
    stage_B_half(Wsum, DM, nbase, kh, tb);
    for (int kb = 0; kb < KHALF; kb += 32) {
      v16bf a  = load_a_frag_f32(X, DM, arow, kh + kb, lane);
      v16bf b0 = load_b_frag_lds(tb, nl0, kb, lane);
      v16bf b1 = load_b_frag_lds(tb, nl1, kb, lane);
      wmma_pair(a, b0, b1, acc0, acc1);
    }
  }
  float bias0 = bq[n0] + bk[n0], bias1 = bq[n1] + bk[n1];
  float v0 = vvec[n0], v1 = vvec[n1];
  int lm = mt * 16 + (lane >> 4) * 8;
#pragma unroll
  for (int r = 0; r < 8; ++r) {
    float p = tanhf(acc0[r] + bias0) * v0 + tanhf(acc1[r] + bias1) * v1;
    atomicAdd(&rowsum[lm + r], p);
  }
  __syncthreads();
  if (threadIdx.x < 64) atomicAdd(&e[mbase + threadIdx.x], rowsum[threadIdx.x]);
}

// softmax over T (axis 0) per column b; in place
__global__ void k_softmax(float* __restrict__ e) {
  __shared__ float sm[128];
  int b = blockIdx.x, t = threadIdx.x;
  float x = e[t * BSZ + b];
  sm[t] = x; __syncthreads();
  for (int s = 64; s > 0; s >>= 1) { if (t < s) sm[t] = fmaxf(sm[t], sm[t + s]); __syncthreads(); }
  float mx = sm[0]; __syncthreads();
  float ex = expf(x - mx);
  sm[t] = ex; __syncthreads();
  for (int s = 64; s > 0; s >>= 1) { if (t < s) sm[t] += sm[t + s]; __syncthreads(); }
  e[t * BSZ + b] = ex * (1.0f / sm[0]);
}

__global__ void k_context(const float* __restrict__ att, const float* __restrict__ X,
                          float* __restrict__ c) {
  int b = blockIdx.x, tid = threadIdx.x;
  int d0 = tid, d1 = tid + 256;
  float s0 = 0.f, s1 = 0.f;
  for (int t = 0; t < TSZ; ++t) {
    float a = att[t * BSZ + b];
    const float* row = X + ((size_t)t * BSZ + b) * DM;
    s0 += a * row[d0];
    s1 += a * row[d1];
  }
  c[b * DM + d0] = s0;
  c[b * DM + d1] = s1;
}

// gC[b,j] = b_ih[j] + c[b,:] . W_ih[j, 512:1024]   (one-time, fp32)
__global__ void k_gc(const float* __restrict__ c, const float* __restrict__ W_ih,
                     const float* __restrict__ b_ih, float* __restrict__ gC) {
  int idx = blockIdx.x * 256 + threadIdx.x;  // 128*1536
  int b = idx / 1536, j = idx % 1536;
  const float* crow = c + b * DM;
  const float* wrow = W_ih + (size_t)j * (2 * DM) + DM;
  float s = b_ih[j];
  for (int k = 0; k < DM; ++k) s += crow[k] * wrow[k];
  gC[idx] = s;
}

// h = inputs[0]; out = emb[SOS]
__global__ void k_init(const float* __restrict__ X, const float* __restrict__ emb,
                       float* __restrict__ h, unsigned short* __restrict__ hbf,
                       unsigned short* __restrict__ outbf) {
  int idx = blockIdx.x * 256 + threadIdx.x;  // 128*512
  int d = idx & (DM - 1);
  float hv = X[idx];
  h[idx] = hv;
  hbf[idx] = f2bf(hv);
  outbf[idx] = f2bf(emb[d]);
}

// ---- per-step kernels ----
// z==0: gi = out @ Wihl^T + gC ; z==1: gh = h @ Whh^T + b_hh
__global__ void __launch_bounds__(256) k_gates(
    const unsigned short* __restrict__ outbf, const unsigned short* __restrict__ hbf,
    const unsigned short* __restrict__ Wihl, const unsigned short* __restrict__ Whh,
    const float* __restrict__ gC, const float* __restrict__ b_hh,
    float* __restrict__ gi, float* __restrict__ gh) {
  __shared__ __align__(16) unsigned short tb[64 * LDB];
  int lane = threadIdx.x & 31, wave = threadIdx.x >> 5;
  int mbase = blockIdx.x * 64, nbase = blockIdx.y * 64;
  int mt = wave & 3, nt = (wave >> 2) * 2;
  int arow = mbase + mt * 16 + (lane & 15);
  int nl0 = nt * 16 + (lane & 15), nl1 = nl0 + 16;
  int n0 = nbase + nl0, n1 = nbase + nl1;
  const unsigned short* A = (blockIdx.z == 0) ? outbf : hbf;
  const unsigned short* W = (blockIdx.z == 0) ? Wihl : Whh;
  v8f acc0 = {}, acc1 = {};
  for (int kh = 0; kh < DM; kh += KHALF) {
    stage_B_half(W, DM, nbase, kh, tb);
    for (int kb = 0; kb < KHALF; kb += 32) {
      v16bf a  = load_a_frag(A, DM, arow, kh + kb, lane);
      v16bf b0 = load_b_frag_lds(tb, nl0, kb, lane);
      v16bf b1 = load_b_frag_lds(tb, nl1, kb, lane);
      wmma_pair(a, b0, b1, acc0, acc1);
    }
  }
  int m = mbase + mt * 16 + (lane >> 4) * 8;
  if (blockIdx.z == 0) {
#pragma unroll
    for (int r = 0; r < 8; ++r) {
      int mm = m + r;
      gi[mm * 1536 + n0] = acc0[r] + gC[mm * 1536 + n0];
      gi[mm * 1536 + n1] = acc1[r] + gC[mm * 1536 + n1];
    }
  } else {
    float bias0 = b_hh[n0], bias1 = b_hh[n1];
#pragma unroll
    for (int r = 0; r < 8; ++r) {
      int mm = m + r;
      gh[mm * 1536 + n0] = acc0[r] + bias0;
      gh[mm * 1536 + n1] = acc1[r] + bias1;
    }
  }
}

__global__ void k_gru_update(const float* __restrict__ gi, const float* __restrict__ gh,
                             float* __restrict__ h, unsigned short* __restrict__ hbf) {
  int idx = blockIdx.x * 256 + threadIdx.x;  // 128*512
  int b = idx >> 9, d = idx & (DM - 1);
  int base = b * 1536;
  float r = 1.0f / (1.0f + expf(-(gi[base + d] + gh[base + d])));
  float z = 1.0f / (1.0f + expf(-(gi[base + 512 + d] + gh[base + 512 + d])));
  float n = tanhf(gi[base + 1024 + d] + r * gh[base + 1024 + d]);
  float hv = (1.0f - z) * n + z * h[idx];
  h[idx] = hv;
  hbf[idx] = f2bf(hv);
}

__global__ void __launch_bounds__(256) k_classifier(
    const unsigned short* __restrict__ hbf, const unsigned short* __restrict__ Wc,
    const float* __restrict__ bc, float* __restrict__ pred) {
  __shared__ __align__(16) unsigned short tb[64 * LDB];
  int lane = threadIdx.x & 31, wave = threadIdx.x >> 5;
  int mbase = blockIdx.x * 64, nbase = blockIdx.y * 64;
  int mt = wave & 3, nt = (wave >> 2) * 2;
  int arow = mbase + mt * 16 + (lane & 15);
  int nl0 = nt * 16 + (lane & 15), nl1 = nl0 + 16;
  int n0 = nbase + nl0, n1 = nbase + nl1;
  v8f acc0 = {}, acc1 = {};
  for (int kh = 0; kh < DM; kh += KHALF) {
    stage_B_half(Wc, DM, nbase, kh, tb);
    for (int kb = 0; kb < KHALF; kb += 32) {
      v16bf a  = load_a_frag(hbf, DM, arow, kh + kb, lane);
      v16bf b0 = load_b_frag_lds(tb, nl0, kb, lane);
      v16bf b1 = load_b_frag_lds(tb, nl1, kb, lane);
      wmma_pair(a, b0, b1, acc0, acc1);
    }
  }
  int m = mbase + mt * 16 + (lane >> 4) * 8;
  float bias0 = bc[n0], bias1 = bc[n1];
#pragma unroll
  for (int r = 0; r < 8; ++r) {
    pred[(size_t)(m + r) * CLS + n0] = acc0[r] + bias0;
    pred[(size_t)(m + r) * CLS + n1] = acc1[r] + bias1;
  }
}

__global__ void k_argmax_embed(const float* __restrict__ pred,
                               const float* __restrict__ emb,
                               unsigned short* __restrict__ outbf) {
  __shared__ float sv[256];
  __shared__ int si[256];
  int b = blockIdx.x, tid = threadIdx.x;
  const float* row = pred + (size_t)b * CLS;
  float best = -3.4e38f; int bi = 0x7fffffff;
  for (int i = tid; i < CLS; i += 256) {
    float vv = row[i];
    if (vv > best) { best = vv; bi = i; }  // keeps lowest index per thread
  }
  sv[tid] = best; si[tid] = bi; __syncthreads();
  for (int s = 128; s > 0; s >>= 1) {
    if (tid < s) {
      float v2 = sv[tid + s]; int i2 = si[tid + s];
      if (v2 > sv[tid] || (v2 == sv[tid] && i2 < si[tid])) { sv[tid] = v2; si[tid] = i2; }
    }
    __syncthreads();
  }
  int top1 = si[0];
  const float* erow = emb + (size_t)top1 * DM;
  outbf[b * DM + tid] = f2bf(erow[tid]);
  outbf[b * DM + tid + 256] = f2bf(erow[tid + 256]);
}

extern "C" void kernel_launch(void* const* d_in, const int* in_sizes, int n_in,
                              void* d_out, int out_size, void* d_ws, size_t ws_size,
                              hipStream_t stream) {
  (void)in_sizes; (void)n_in; (void)out_size; (void)ws_size;
  const float* inputs = (const float*)d_in[0];
  const float* Wq  = (const float*)d_in[1];
  const float* bq  = (const float*)d_in[2];
  const float* Wk  = (const float*)d_in[3];
  const float* bk  = (const float*)d_in[4];
  const float* vvec= (const float*)d_in[5];
  const float* W_ih= (const float*)d_in[6];
  const float* b_ih= (const float*)d_in[7];
  const float* W_hh= (const float*)d_in[8];
  const float* b_hh= (const float*)d_in[9];
  const float* emb = (const float*)d_in[10];
  const float* Wc  = (const float*)d_in[11];
  const float* bc  = (const float*)d_in[12];
  float* out = (float*)d_out;

  char* ws = (char*)d_ws;
  size_t off = 0;
  auto alloc = [&](size_t bytes) { size_t o = off; off += (bytes + 255) & ~(size_t)255; return o; };
  unsigned short* wsum_bf = (unsigned short*)(ws + alloc((size_t)DM * DM * 2));
  unsigned short* wihl_bf = (unsigned short*)(ws + alloc((size_t)1536 * DM * 2));
  unsigned short* whh_bf  = (unsigned short*)(ws + alloc((size_t)1536 * DM * 2));
  unsigned short* wc_bf   = (unsigned short*)(ws + alloc((size_t)CLS * DM * 2));
  float* e_att = (float*)(ws + alloc((size_t)TSZ * BSZ * 4));
  float* cctx  = (float*)(ws + alloc((size_t)BSZ * DM * 4));
  float* gC    = (float*)(ws + alloc((size_t)BSZ * 1536 * 4));
  float* gi    = (float*)(ws + alloc((size_t)BSZ * 1536 * 4));
  float* gh    = (float*)(ws + alloc((size_t)BSZ * 1536 * 4));
  float* hbuf  = (float*)(ws + alloc((size_t)BSZ * DM * 4));
  unsigned short* hbf   = (unsigned short*)(ws + alloc((size_t)BSZ * DM * 2));
  unsigned short* outbf = (unsigned short*)(ws + alloc((size_t)BSZ * DM * 2));

  // one-time: weight conversions (PyTorch [N,K] row-major -> bf16, no transpose)
  k_wsum<<<1024, 256, 0, stream>>>(wsum_bf, Wq, Wk);
  k_convert<<<1024, 256, 0, stream>>>(wihl_bf, W_ih, 1536, 1024, 512);
  k_convert<<<1024, 256, 0, stream>>>(whh_bf, W_hh, 1536, 512, 512);
  k_convert<<<2048, 256, 0, stream>>>(wc_bf, Wc, CLS, 512, 512);

  // attention scores (loop-invariant)
  hipMemsetAsync(e_att, 0, (size_t)TSZ * BSZ * 4, stream);
  k_att_gemm<<<dim3(256, 8), 256, 0, stream>>>(inputs, wsum_bf, bq, bk, vvec, e_att);
  k_softmax<<<128, 128, 0, stream>>>(e_att);
  k_context<<<128, 256, 0, stream>>>(e_att, inputs, cctx);
  k_gc<<<768, 256, 0, stream>>>(cctx, W_ih, b_ih, gC);
  k_init<<<256, 256, 0, stream>>>(inputs, emb, hbuf, hbf, outbf);

  // serial greedy decode
  for (int s = 0; s < MAXLEN; ++s) {
    float* pred = out + (size_t)s * BSZ * CLS;
    k_gates<<<dim3(2, 24, 2), 256, 0, stream>>>(outbf, hbf, wihl_bf, whh_bf, gC, b_hh, gi, gh);
    k_gru_update<<<256, 256, 0, stream>>>(gi, gh, hbuf, hbf);
    k_classifier<<<dim3(2, 125), 256, 0, stream>>>(hbf, wc_bf, bc, pred);
    k_argmax_embed<<<128, 256, 0, stream>>>(pred, emb, outbf);
  }
}
